// MultiHeadSelfAttention_25013889532557
// MI455X (gfx1250) — compile-verified, matrix operand order checked by reading, not run on hardware
//
#include <hip/hip_runtime.h>

// ---------------------------------------------------------------------------
// MHA forward for MI455X (gfx1250, wave32, WMMA 16x16x32 bf16)
// B=4, S=2048, D=1024, H=16, DH=64
// Round 4: fix async-to-LDS builtin arg types (int4 __device__* / __shared__*).
// Block-cooperative LDS-double-buffered GEMMs + pipelined flash attention.
// ---------------------------------------------------------------------------

constexpr int BATCH  = 4;
constexpr int SEQ    = 2048;
constexpr int DMODEL = 1024;
constexpr int NH     = 16;
constexpr int DHEAD  = 64;
constexpr int MTOT   = BATCH * SEQ;          // 8192 rows for the projections

constexpr int LDS_RS   = 72;                 // LDS row pitch (bf16): 16 frag rows
                                             // -> 16 distinct banks, no conflicts
constexpr int TILE_ELT = 64 * LDS_RS;        // one 64(n) x 64(k) staged tile

typedef __bf16 bf16;
typedef bf16  v16bf __attribute__((ext_vector_type(16)));
typedef bf16  v8bf  __attribute__((ext_vector_type(8)));
typedef bf16  v4bf  __attribute__((ext_vector_type(4)));
typedef float v8f   __attribute__((ext_vector_type(8)));
typedef int   v4i   __attribute__((ext_vector_type(4)));

// gfx1250 async global->LDS path (ASYNCcnt) when the toolchain exposes it.
#if defined(__AMDGCN__) && \
    __has_builtin(__builtin_amdgcn_global_load_async_to_lds_b128) && \
    __has_builtin(__builtin_amdgcn_s_wait_asynccnt)
#define ASYNC_LDS 1
typedef __attribute__((address_space(1))) v4i* gv4ip;   // global int4*
typedef __attribute__((address_space(3))) v4i* lv4ip;   // LDS int4*
#else
#define ASYNC_LDS 0
#endif

// ---- WMMA wrapper: D = A*B + C, f32 accum, bf16 operands --------------------
__device__ __forceinline__ v8f wmma_bf16(v16bf a, v16bf b, v8f c) {
    return __builtin_amdgcn_wmma_f32_16x16x32_bf16(
        /*neg_a=*/false, a, /*neg_b=*/false, b,
        /*c_mod=*/(short)0, c, /*reuse_a=*/false, /*reuse_b=*/false);
}

// ---- Fragments --------------------------------------------------------------
// CDNA5 16-bit A/B layout: lane L (L<16) holds row (L%16) with K={0..7,16..23};
// lane L>=16 holds the same rows with K={8..15,24..31}.  Caller passes
// p = row_ptr + k0 + (half ? 8 : 0); fragment = two 16B chunks 32B apart.
struct Frag { v8bf lo, hi; };

__device__ __forceinline__ Frag load_raw(const bf16* p) {
    Frag f;
    f.lo = *(const v8bf*)(p);
    f.hi = *(const v8bf*)(p + 16);
    return f;
}
__device__ __forceinline__ v16bf cat(Frag f) {
    return __builtin_shufflevector(f.lo, f.hi,
        0, 1, 2, 3, 4, 5, 6, 7, 8, 9, 10, 11, 12, 13, 14, 15);
}

__device__ __forceinline__ void zero_acc(v8f& a) {
#pragma unroll
    for (int i = 0; i < 8; ++i) a[i] = 0.0f;
}

// 16-lane reductions (D-matrix rows live across lanes 0..15 / 16..31; xor
// masks 1,2,4,8 never cross the 16-lane half boundary on wave32).
__device__ __forceinline__ float redmax16(float x) {
#pragma unroll
    for (int off = 1; off < 16; off <<= 1) x = fmaxf(x, __shfl_xor(x, off, 32));
    return x;
}
__device__ __forceinline__ float redsum16(float x) {
#pragma unroll
    for (int off = 1; off < 16; off <<= 1) x += __shfl_xor(x, off, 32);
    return x;
}

// ---------------------------------------------------------------------------
// Kernel 1: fp32 -> bf16 cast (vectorized x4)
// ---------------------------------------------------------------------------
__global__ void __launch_bounds__(256)
cast_f32_bf16(const float* __restrict__ s, bf16* __restrict__ d, int n4) {
    int i = blockIdx.x * blockDim.x + threadIdx.x;
    if (i < n4) {
        float4 f = ((const float4*)s)[i];
        v4bf o;
        o[0] = (bf16)f.x; o[1] = (bf16)f.y; o[2] = (bf16)f.z; o[3] = (bf16)f.w;
        *(v4bf*)(d + (size_t)i * 4) = o;
    }
}

// ---------------------------------------------------------------------------
// Cooperative staging of one 64(n) x 64(k) weight tile into LDS.
// 256 threads: thread covers rows n = tid/8 and n+32, cols kk = (tid%8)*8.
// ---------------------------------------------------------------------------
#if ASYNC_LDS
__device__ __forceinline__ void stage_issue(bf16* dst, const bf16* __restrict__ src,
                                            int tid) {
    int n  = tid >> 3;
    int kk = (tid & 7) * 8;
    const bf16* s0 = src + (size_t)n * DMODEL + kk;
    const bf16* s1 = src + (size_t)(n + 32) * DMODEL + kk;
    bf16* d0 = dst + n * LDS_RS + kk;
    bf16* d1 = dst + (n + 32) * LDS_RS + kk;
    __builtin_amdgcn_global_load_async_to_lds_b128(
        (gv4ip)(void*)s0, (lv4ip)(void*)d0, 0, 0);
    __builtin_amdgcn_global_load_async_to_lds_b128(
        (gv4ip)(void*)s1, (lv4ip)(void*)d1, 0, 0);
}
__device__ __forceinline__ void stage_wait() {
    __builtin_amdgcn_s_wait_asynccnt(0);
}
#else
struct StageReg { v8bf c0, c1; };
__device__ __forceinline__ StageReg stage_load(const bf16* __restrict__ src,
                                               int tid) {
    int n  = tid >> 3;
    int kk = (tid & 7) * 8;
    StageReg r;
    r.c0 = *(const v8bf*)(src + (size_t)n * DMODEL + kk);
    r.c1 = *(const v8bf*)(src + (size_t)(n + 32) * DMODEL + kk);
    return r;
}
__device__ __forceinline__ void stage_store(bf16* dst, StageReg r, int tid) {
    int n  = tid >> 3;
    int kk = (tid & 7) * 8;
    *(v8bf*)(dst + n * LDS_RS + kk)        = r.c0;
    *(v8bf*)(dst + (n + 32) * LDS_RS + kk) = r.c1;
}
#endif

// ---------------------------------------------------------------------------
// Block-cooperative GEMM core.  Block = 8 waves; block tile M=128 x N=64;
// wave w owns rows m0..m0+15.  W tile is double-buffered in LDS; stage of
// tile i+1 is issued before computing on tile i (one barrier per K-step).
// A fragments are register double-buffered (clamped prefetch).
// ---------------------------------------------------------------------------
__device__ __forceinline__ void gemm_block_16x64(const bf16* __restrict__ arow,
                                                 const bf16* __restrict__ wbase,
                                                 bf16* ldsW, int tid, int row,
                                                 int hoff, v8f acc[4]) {
#if ASYNC_LDS
    stage_issue(ldsW, wbase, tid);
    stage_wait();
#else
    stage_store(ldsW, stage_load(wbase, tid), tid);
#endif
    __syncthreads();

    Frag fa0 = load_raw(arow);
    Frag fa1 = load_raw(arow + 32);

    int buf = 0;
    for (int kt = 0; kt < DMODEL; kt += 64) {
        bool more = (kt + 64) < DMODEL;
        int  kn   = more ? kt + 64 : kt;        // clamped prefetch addresses
        // ---- issue stage of next W tile (hides under compute) -------------
#if ASYNC_LDS
        if (more) stage_issue(ldsW + (buf ^ 1) * TILE_ELT, wbase + kn, tid);
#else
        StageReg st = stage_load(wbase + kn, tid);
#endif
        // ---- prefetch next A fragments -------------------------------------
        Frag na0 = load_raw(arow + kn);
        Frag na1 = load_raw(arow + kn + 32);

        // ---- compute on current tile: 2 K-substeps x 4 N-tiles ------------
        const bf16* lw = ldsW + buf * TILE_ELT + row * LDS_RS + hoff;
        v16bf a0 = cat(fa0), a1 = cat(fa1);
#pragma unroll
        for (int t = 0; t < 4; ++t) {
            acc[t] = wmma_bf16(a0, cat(load_raw(lw + t * 16 * LDS_RS)), acc[t]);
            acc[t] = wmma_bf16(a1, cat(load_raw(lw + t * 16 * LDS_RS + 32)), acc[t]);
        }

        // ---- commit the staged tile and flip buffers ----------------------
#if ASYNC_LDS
        stage_wait();
#else
        if (more) stage_store(ldsW + (buf ^ 1) * TILE_ELT, st, tid);
#endif
        __syncthreads();
        buf ^= 1;
        fa0 = na0; fa1 = na1;
    }
}

// ---------------------------------------------------------------------------
// Kernel 2: fused QKV projection.  grid.y selects which = {Q, K, V}.
// Q,K stored [B,H,S,DH]; V stored transposed [B,H,DH,S] so PV's B-operand in
// the attention kernel becomes contiguous row loads (packed b128 store here).
// ---------------------------------------------------------------------------
__global__ void __launch_bounds__(256)
qkv_proj_kernel(const bf16* __restrict__ xb,
                const bf16* __restrict__ wq, const bf16* __restrict__ wk,
                const bf16* __restrict__ wv,
                bf16* __restrict__ qout, bf16* __restrict__ kout,
                bf16* __restrict__ vtout) {
    __shared__ bf16 ldsW[2 * TILE_ELT];      // 18 KB, double-buffered W tile

    const int tid  = threadIdx.x;
    const int lane = tid & 31;
    const int wid  = tid >> 5;
    const int half = lane >> 4;
    const int row  = lane & 15;
    const int hoff = half ? 8 : 0;
    const int which = blockIdx.y;

    int bm = blockIdx.x >> 4;                // 64 M-blocks of 128 rows
    int nc = blockIdx.x & 15;                // 16 N-chunks of 64 cols (1 head)
    int m0 = bm * 128 + wid * 16;
    int n0 = nc * 64;

    const bf16* w = (which == 0) ? wq : (which == 1) ? wk : wv;

    v8f acc[4];
#pragma unroll
    for (int t = 0; t < 4; ++t) zero_acc(acc[t]);

    const bf16* arow = xb + (size_t)(m0 + row) * DMODEL + hoff;
    gemm_block_16x64(arow, w + (size_t)n0 * DMODEL, ldsW, tid, row, hoff, acc);

    // D layout: lane L, acc[t][v] = element (m = (half ? v+8 : v), n = row)
    int b     = m0 / SEQ;
    int sbase = m0 % SEQ;
    int h     = nc;                           // each 64-col chunk == one head
    if (which == 0) {                         // Q, scaled by 1/sqrt(DHEAD)
#pragma unroll
        for (int t = 0; t < 4; ++t)
#pragma unroll
            for (int v = 0; v < 8; ++v) {
                int m = half ? v + 8 : v;
                qout[((size_t)(b * NH + h) * SEQ + sbase + m) * DHEAD +
                     t * 16 + row] = (bf16)(acc[t][v] * 0.125f);
            }
    } else if (which == 1) {                  // K
#pragma unroll
        for (int t = 0; t < 4; ++t)
#pragma unroll
            for (int v = 0; v < 8; ++v) {
                int m = half ? v + 8 : v;
                kout[((size_t)(b * NH + h) * SEQ + sbase + m) * DHEAD +
                     t * 16 + row] = (bf16)acc[t][v];
            }
    } else {                                  // V -> Vt [B,H,DH,S], packed b128
#pragma unroll
        for (int t = 0; t < 4; ++t) {
            v8bf pk;
#pragma unroll
            for (int v = 0; v < 8; ++v) pk[v] = (bf16)acc[t][v];
            int dh = t * 16 + row;
            size_t base = ((size_t)(b * NH + h) * DHEAD + dh) * SEQ +
                          sbase + hoff;
            *(v8bf*)(vtout + base) = pk;
        }
    }
}

// ---------------------------------------------------------------------------
// Kernel 3: causal flash attention.  One wave per 16-query tile; 32 keys per
// step; online softmax.  V fragments and the NEXT step's K fragments are
// issued before the softmax so their latency hides under the VALU/TRANS chain
// and the LDS P round-trip (D-layout -> A-layout staging, 1KB per wave).
// ---------------------------------------------------------------------------
__global__ void __launch_bounds__(256)
attn_kernel(const bf16* __restrict__ qb, const bf16* __restrict__ kb,
            const bf16* __restrict__ vt, bf16* __restrict__ ctx) {
    __shared__ bf16 plds[8][16 * 32];     // per-wave P tile [16 q][32 k]

    const int lane = threadIdx.x & 31;
    const int wid  = threadIdx.x >> 5;
    const int half = lane >> 4;
    const int row  = lane & 15;
    const int hoff = half ? 8 : 0;

    int job = blockIdx.x * 8 + wid;       // 64 bh * 128 qtiles = 8192
    int bh = job >> 7;
    int qt = job & 127;
    int q0 = qt * 16;

    // Q fragments (dh 0..31 and 32..63), reused for the whole key loop
    const bf16* qrow = qb + ((size_t)bh * SEQ + q0 + row) * DHEAD + hoff;
    v16bf aq0 = cat(load_raw(qrow));
    v16bf aq1 = cat(load_raw(qrow + 32));

    const bf16* krow = kb + ((size_t)bh * SEQ + row) * DHEAD + hoff;
    const bf16* vrow = vt + ((size_t)bh * DHEAD + row) * SEQ + hoff;

    float mst[8], lst[8];
    v8f oacc[4];
#pragma unroll
    for (int v = 0; v < 8; ++v) { mst[v] = -1e30f; lst[v] = 0.0f; }
#pragma unroll
    for (int t = 0; t < 4; ++t) zero_acc(oacc[t]);

    const int kend = q0 + 16;             // causal: keys 0 .. q0+15

    // K fragments for step 0 (rows k0+{0..15} and k0+{16..31}, dh chunks 0/1)
    Frag fk[4];
    fk[0] = load_raw(krow);
    fk[1] = load_raw(krow + 32);
    fk[2] = load_raw(krow + 16 * DHEAD);
    fk[3] = load_raw(krow + 16 * DHEAD + 32);

    for (int k0 = 0; k0 < kend; k0 += 32) {
        // ---- scores: S[16x32] = Q(16x64) * K_j^T --------------------------
        v8f sc0, sc1; zero_acc(sc0); zero_acc(sc1);
        sc0 = wmma_bf16(aq0, cat(fk[0]), sc0);
        sc0 = wmma_bf16(aq1, cat(fk[1]), sc0);
        sc1 = wmma_bf16(aq0, cat(fk[2]), sc1);
        sc1 = wmma_bf16(aq1, cat(fk[3]), sc1);

        // ---- issue V loads now (consumed after softmax) -------------------
        Frag fv[4];
#pragma unroll
        for (int t = 0; t < 4; ++t)
            fv[t] = load_raw(vrow + (size_t)t * 16 * SEQ + k0);

        // ---- prefetch next step's K fragments (clamped at the tail) -------
        int kn = (k0 + 32 < kend) ? (k0 + 32) : k0;
        Frag nk[4];
        nk[0] = load_raw(krow + (size_t)kn * DHEAD);
        nk[1] = load_raw(krow + (size_t)kn * DHEAD + 32);
        nk[2] = load_raw(krow + (size_t)(kn + 16) * DHEAD);
        nk[3] = load_raw(krow + (size_t)(kn + 16) * DHEAD + 32);

        // ---- causal mask + online softmax ---------------------------------
#pragma unroll
        for (int v = 0; v < 8; ++v) {
            int m = half ? v + 8 : v;
            int qpos = q0 + m;
            float s0 = (k0 + row      <= qpos) ? sc0[v] : -1e30f;
            float s1 = (k0 + 16 + row <= qpos) ? sc1[v] : -1e30f;
            float rm   = redmax16(fmaxf(s0, s1));
            float mnew = fmaxf(mst[v], rm);
            float alpha = __expf(mst[v] - mnew);
            mst[v] = mnew;
            float p0 = __expf(s0 - mnew);
            float p1 = __expf(s1 - mnew);
            lst[v] = lst[v] * alpha + redsum16(p0 + p1);
#pragma unroll
            for (int t = 0; t < 4; ++t) oacc[t][v] *= alpha;
            // stage P to LDS in [q][k] row-major (bf16)
            plds[wid][m * 32 + row]      = (bf16)p0;
            plds[wid][m * 32 + 16 + row] = (bf16)p1;
        }

        // ---- reload P as an A-fragment, then O += P * V -------------------
        v16bf pa = cat(load_raw(&plds[wid][row * 32 + hoff]));
#pragma unroll
        for (int t = 0; t < 4; ++t)
            oacc[t] = wmma_bf16(pa, cat(fv[t]), oacc[t]);

#pragma unroll
        for (int t = 0; t < 4; ++t) fk[t] = nk[t];
    }

    // ---- normalize + store ctx [B,S,D] (heads concatenated) ---------------
    int b = bh >> 4, h = bh & 15;
#pragma unroll
    for (int v = 0; v < 8; ++v) {
        int m = half ? v + 8 : v;
        float inv = 1.0f / lst[v];
        size_t base = ((size_t)b * SEQ + q0 + m) * DMODEL + h * DHEAD;
#pragma unroll
        for (int t = 0; t < 4; ++t)
            ctx[base + t * 16 + row] = (bf16)(oacc[t][v] * inv);
    }
}

// ---------------------------------------------------------------------------
// Kernel 4: output projection, fp32 result.  out = ctx * Wo^T
// ---------------------------------------------------------------------------
__global__ void __launch_bounds__(256)
oproj_kernel(const bf16* __restrict__ ctx, const bf16* __restrict__ wo,
             float* __restrict__ out) {
    __shared__ bf16 ldsW[2 * TILE_ELT];

    const int tid  = threadIdx.x;
    const int lane = tid & 31;
    const int wid  = tid >> 5;
    const int half = lane >> 4;
    const int row  = lane & 15;
    const int hoff = half ? 8 : 0;

    int bm = blockIdx.x >> 4;
    int nc = blockIdx.x & 15;
    int m0 = bm * 128 + wid * 16;
    int n0 = nc * 64;

    v8f acc[4];
#pragma unroll
    for (int t = 0; t < 4; ++t) zero_acc(acc[t]);

    const bf16* arow = ctx + (size_t)(m0 + row) * DMODEL + hoff;
    gemm_block_16x64(arow, wo + (size_t)n0 * DMODEL, ldsW, tid, row, hoff, acc);

#pragma unroll
    for (int t = 0; t < 4; ++t)
#pragma unroll
        for (int v = 0; v < 8; ++v) {
            int m = m0 + (half ? v + 8 : v);
            out[(size_t)m * DMODEL + n0 + t * 16 + row] = acc[t][v];
        }
}

// ---------------------------------------------------------------------------
// Launcher
// ---------------------------------------------------------------------------
extern "C" void kernel_launch(void* const* d_in, const int* in_sizes, int n_in,
                              void* d_out, int out_size, void* d_ws, size_t ws_size,
                              hipStream_t stream) {
    const float* x  = (const float*)d_in[0];
    const float* wq = (const float*)d_in[1];
    const float* wk = (const float*)d_in[2];
    const float* wv = (const float*)d_in[3];
    const float* wo = (const float*)d_in[4];
    float* out = (float*)d_out;

    // workspace carve-up (all sizes 256B aligned by construction)
    char* p = (char*)d_ws;
    size_t off = 0;
    auto alloc = [&](size_t bytes) { void* r = p + off; off += bytes; return r; };
    const size_t nX = (size_t)MTOT * DMODEL;       // 8388608
    const size_t nW = (size_t)DMODEL * DMODEL;     // 1048576
    bf16* xb  = (bf16*)alloc(nX * 2);
    bf16* wqb = (bf16*)alloc(nW * 2);
    bf16* wkb = (bf16*)alloc(nW * 2);
    bf16* wvb = (bf16*)alloc(nW * 2);
    bf16* wob = (bf16*)alloc(nW * 2);
    bf16* qb  = (bf16*)alloc(nX * 2);              // [B,H,S,DH]
    bf16* kb  = (bf16*)alloc(nX * 2);              // [B,H,S,DH]
    bf16* vtb = (bf16*)alloc(nX * 2);              // [B,H,DH,S]
    bf16* ctb = (bf16*)alloc(nX * 2);              // [B,S,D]
    (void)ws_size;

    // 1) casts
    {
        int n4 = (int)(nX / 4);
        cast_f32_bf16<<<(n4 + 255) / 256, 256, 0, stream>>>(x, xb, n4);
        int w4 = (int)(nW / 4);
        cast_f32_bf16<<<(w4 + 255) / 256, 256, 0, stream>>>(wq, wqb, w4);
        cast_f32_bf16<<<(w4 + 255) / 256, 256, 0, stream>>>(wk, wkb, w4);
        cast_f32_bf16<<<(w4 + 255) / 256, 256, 0, stream>>>(wv, wvb, w4);
        cast_f32_bf16<<<(w4 + 255) / 256, 256, 0, stream>>>(wo, wob, w4);
    }
    // 2) fused QKV projection: 64 M-blocks x 16 N-chunks, 8 waves/block
    qkv_proj_kernel<<<dim3(1024, 3), 256, 0, stream>>>(xb, wqb, wkb, wvb,
                                                       qb, kb, vtb);
    // 3) flash attention: 8192 wave-jobs
    attn_kernel<<<1024, 256, 0, stream>>>(qb, kb, vtb, ctb);
    // 4) output projection
    oproj_kernel<<<1024, 256, 0, stream>>>(ctb, wob, out);
}
